// TransformerBlock_43095701848101
// MI455X (gfx1250) — compile-verified, matrix operand order checked by reading, not run on hardware
//
#include <hip/hip_runtime.h>
#include <hip/hip_bf16.h>

// ---------------------------------------------------------------------------
// Transformer block for MI455X (gfx1250, wave32, WMMA).
// All large matmuls run through v_wmma_f32_16x16x32_bf16 (f32 accumulate).
// GEMMs are double-buffered in LDS (global loads for step s+1 issued before
// the WMMAs of step s) so HBM latency hides behind the matrix pipe.
// ---------------------------------------------------------------------------

typedef __bf16 bf16;
typedef __attribute__((ext_vector_type(16))) __bf16 v16bf;
typedef __attribute__((ext_vector_type(8)))  __bf16 v8bf;
typedef __attribute__((ext_vector_type(8)))  float  v8f;

#define Bb 128
#define Tt 256
#define Cc 384
#define Hh 6
#define Dd 64
#define FFf 1536
#define Mm (Bb * Tt)           // 32768 tokens

#define SCALE 0.05103103630798287f   // C ** -0.5, C = 384
#define NEG_INF (-__builtin_inff())

// float -> bf16, round-to-nearest-even via integer trick
__device__ inline bf16 f2bf(float f) {
    union { float f; unsigned u; } in; in.f = f;
    unsigned r = (in.u + 0x7FFFu + ((in.u >> 16) & 1u)) >> 16;
    union { unsigned short s; bf16 b; } out; out.s = (unsigned short)r;
    return out.b;
}

// Load a 16x32 bf16 A/B fragment for v_wmma_f32_16x16x32_bf16 from LDS.
// Per-lane layout (CDNA5 ISA 7.12.2): lane L holds row (L%16); halves 0..7 at
// K = klo..klo+7, halves 8..15 at K = klo+16..klo+23, klo = (L<16 ? 0 : 8).
// Caller passes p = &row[klo]; two contiguous 16B loads (ds_load_b128).
__device__ inline v16bf ldfrag(const bf16* p) {
    union { v16bf v; v8bf h[2]; } u;
    u.h[0] = *(const v8bf*)(p);
    u.h[1] = *(const v8bf*)(p + 16);
    return u.v;
}

__device__ inline v8f wmma_bf16(v16bf a, v16bf b, v8f c) {
    return __builtin_amdgcn_wmma_f32_16x16x32_bf16(
        /*neg_a=*/false, a, /*neg_b=*/false, b,
        /*c_mod=*/(short)0, c, /*reuse_a=*/false, /*reuse_b=*/false);
}

// ---------------------------------------------------------------------------
// Elementwise converters / weight packers
// ---------------------------------------------------------------------------
__global__ void cvt_f32_bf16_kernel(const float* __restrict__ in,
                                    bf16* __restrict__ out, int n) {
    for (int i = blockIdx.x * blockDim.x + threadIdx.x; i < n;
         i += gridDim.x * blockDim.x)
        out[i] = f2bf(in[i]);
}

// (H, C, D) fp32  ->  [C, H*D] bf16  (per-head QKV weights become one GEMM B)
__global__ void pack_hcd_kernel(const float* __restrict__ in,
                                bf16* __restrict__ out) {
    const int n = Hh * Cc * Dd;
    for (int i = blockIdx.x * blockDim.x + threadIdx.x; i < n;
         i += gridDim.x * blockDim.x) {
        int h = i / (Cc * Dd);
        int r = i % (Cc * Dd);
        int c = r / Dd;
        int d = r % Dd;
        out[c * (Hh * Dd) + h * Dd + d] = f2bf(in[i]);
    }
}

// ---------------------------------------------------------------------------
// Generic bf16 GEMM: C[M,N] = A[M,K] * B[K,N] + bias, optional ReLU.
// Block tile 128x64, K-step 32, 8 waves; wave w owns rows w*16..w*16+15 and
// all 4 column tiles (A-fragment reuse). B staged transposed in LDS so
// fragment loads are contiguous b128s. Double-buffered LDS: globals for the
// next k-step are register-staged before this step's WMMAs, one barrier/step.
// ---------------------------------------------------------------------------
#define GBM 128
#define GBN 64
#define GBK 32
#define LDA 40   // padded LDS stride in halves (80B, keeps 16B alignment)

template <int RELU, int OUT_F32, int OUT_BF16>
__global__ __launch_bounds__(256) void gemm_bf16_kernel(
    const bf16* __restrict__ A, const bf16* __restrict__ B,
    const float* __restrict__ bias,
    float* __restrict__ Cf, bf16* __restrict__ Cb,
    int M, int N, int K)
{
    __shared__ bf16 Alds[2][GBM * LDA];   // [buf][row][k]
    __shared__ bf16 Blds[2][GBN * LDA];   // [buf][n][k] (transposed)

    const int tid  = threadIdx.x;
    const int lane = tid & 31;
    const int w    = tid >> 5;
    const int l16  = lane & 15;
    const int klo  = (lane < 16) ? 0 : 8;
    const int blockM = blockIdx.y * GBM;
    const int blockN = blockIdx.x * GBN;

    // staging assignments
    const int rA  = tid >> 1;               // A row 0..127
    const int cA  = (tid & 1) * 16;         // A half-offset 0 or 16
    const int kB  = tid >> 3;               // B k-row 0..31
    const int nB0 = (tid & 7) * 8;          // B n-offset (8 contiguous n)
    const bf16* aSrc = A + (size_t)(blockM + rA) * K + cA;
    const bf16* bSrc = B + (size_t)kB * N + blockN + nB0;

    uint4 aR0, aR1, bR;
    auto gload = [&](int k0) {
        aR0 = *(const uint4*)(aSrc + k0);
        aR1 = *(const uint4*)(aSrc + k0 + 8);
        bR  = *(const uint4*)(bSrc + (size_t)k0 * N);
    };
    auto sstore = [&](int buf) {
        *(uint4*)&Alds[buf][rA * LDA + cA]     = aR0;
        *(uint4*)&Alds[buf][rA * LDA + cA + 8] = aR1;
        const bf16* bh = (const bf16*)&bR;
        #pragma unroll
        for (int j = 0; j < 8; ++j)
            Blds[buf][(nB0 + j) * LDA + kB] = bh[j];
    };

    v8f acc[4] = {};

    gload(0);
    sstore(0);
    __syncthreads();

    const int nsteps = K / GBK;
    for (int s = 0; s < nsteps; ++s) {
        const int cur = s & 1;
        if (s + 1 < nsteps) gload((s + 1) * GBK);   // issue next tile's loads

        v16bf a = ldfrag(&Alds[cur][(w * 16 + l16) * LDA + klo]);
        #pragma unroll
        for (int nt = 0; nt < 4; ++nt) {
            v16bf b = ldfrag(&Blds[cur][(nt * 16 + l16) * LDA + klo]);
            acc[nt] = wmma_bf16(a, b, acc[nt]);
        }

        if (s + 1 < nsteps) sstore(1 - cur);        // fill the other buffer
        __syncthreads();
    }

    // --- epilogue: C/D layout => VGPR r holds row r (lanes 0-15) / r+8 ---
    const int mBase = blockM + w * 16 + klo;
    #pragma unroll
    for (int nt = 0; nt < 4; ++nt) {
        int n = blockN + nt * 16 + l16;
        float bv = bias[n];
        #pragma unroll
        for (int r = 0; r < 8; ++r) {
            float v = acc[nt][r] + bv;
            if (RELU) v = v > 0.0f ? v : 0.0f;
            size_t idx = (size_t)(mBase + r) * N + n;
            if (OUT_F32)  Cf[idx] = v;
            if (OUT_BF16) Cb[idx] = f2bf(v);
        }
    }
}

// ---------------------------------------------------------------------------
// Fused attention: one workgroup per (b,h). scores = K Q^T * scale (the
// reference's K-row/Q-col quirk), causal mask (keep s <= t), softmax over s,
// O = P V. Fully-masked 16x16 tiles are skipped (halves the WMMA work).
// q/k/v inputs are bf16 in [B, T, H*D] layout (n = h*64+d) from the QKV GEMM.
// ---------------------------------------------------------------------------
#define LKQ 72     // padded LDS stride for K/Q rows (halves)
#define LVP 264    // padded LDS stride for Vt / P rows (halves)

__global__ __launch_bounds__(256) void attn_kernel(
    const bf16* __restrict__ qb, const bf16* __restrict__ kb,
    const bf16* __restrict__ vb, bf16* __restrict__ attnb)
{
    extern __shared__ char smem[];
    bf16*  Kl = (bf16*)smem;                 // [256][LKQ]
    bf16*  Ql = Kl + Tt * LKQ;               // [256][LKQ]
    bf16*  Vt = Ql + Tt * LKQ;               // transposed V: [64][LVP]
    float* S  = (float*)(Vt + Dd * LVP);     // per-wave [16][256] f32
    bf16*  P  = (bf16*)(S + 8 * 16 * Tt);    // per-wave [16][LVP] bf16

    const int tid  = threadIdx.x;
    const int lane = tid & 31;
    const int w    = tid >> 5;
    const int l16  = lane & 15;
    const int klo  = (lane < 16) ? 0 : 8;

    const int bh = blockIdx.x;
    const int b  = bh / Hh;
    const int h  = bh % Hh;
    const size_t base = ((size_t)b * Tt) * Cc + (size_t)h * Dd;  // elem offset

    // --- stage K and Q rows (8-half vector copies) ---
    for (int idx = tid; idx < Tt * 8; idx += 256) {
        int t  = idx >> 3;
        int c8 = (idx & 7) * 8;
        *(uint4*)&Kl[t * LKQ + c8] = *(const uint4*)(kb + base + (size_t)t * Cc + c8);
        *(uint4*)&Ql[t * LKQ + c8] = *(const uint4*)(qb + base + (size_t)t * Cc + c8);
    }
    // --- stage V transposed: Vt[d][s] = V[s][d] ---
    for (int e = tid; e < Tt * Dd; e += 256) {
        int s = e >> 6;
        int d = e & 63;
        Vt[d * LVP + s] = vb[base + (size_t)s * Cc + d];
    }
    __syncthreads();

    float* Sw = S + w * 16 * Tt;
    bf16*  Pw = P + w * 16 * LVP;

    for (int rb = 0; rb < 2; ++rb) {
        const int t0 = rb * 128 + w * 16;   // this wave's 16 key rows

        // A-fragments from K rows (d-slices 0..31 and 32..63), reused over s
        v16bf aK0 = ldfrag(&Kl[(t0 + l16) * LKQ + 0  + klo]);
        v16bf aK1 = ldfrag(&Kl[(t0 + l16) * LKQ + 32 + klo]);

        // ---- scores: S[t][s] = sum_d K[t][d] Q[s][d]  (only s-tiles <= t0)
        const int nTiles = t0 / 16 + 1;
        for (int st = 0; st < nTiles; ++st) {
            int s0 = st * 16;
            v16bf bQ0 = ldfrag(&Ql[(s0 + l16) * LKQ + 0  + klo]);
            v16bf bQ1 = ldfrag(&Ql[(s0 + l16) * LKQ + 32 + klo]);
            v8f accS = {};
            accS = wmma_bf16(aK0, bQ0, accS);
            accS = wmma_bf16(aK1, bQ1, accS);
            int s = s0 + l16;
            #pragma unroll
            for (int r = 0; r < 8; ++r) {
                int rl = r + klo;                    // local row 0..15
                int t  = t0 + rl;
                float v = accS[r] * SCALE;
                if (s > t) v = NEG_INF;              // causal mask
                Sw[rl * Tt + s] = v;
            }
        }

        // ---- softmax over s (valid region s <= t), write P as bf16 ----
        const int sPad = (t0 & ~31) + 32;            // pad to WMMA K multiple
        for (int r = 0; r < 16; ++r) {
            int t = t0 + r;
            float m = NEG_INF;
            for (int s = lane; s <= t; s += 32) m = fmaxf(m, Sw[r * Tt + s]);
            #pragma unroll
            for (int off = 16; off; off >>= 1) m = fmaxf(m, __shfl_xor(m, off, 32));
            float sum = 0.0f;
            for (int s = lane; s <= t; s += 32) {
                float e = __expf(Sw[r * Tt + s] - m);
                Sw[r * Tt + s] = e;
                sum += e;
            }
            #pragma unroll
            for (int off = 16; off; off >>= 1) sum += __shfl_xor(sum, off, 32);
            float inv = 1.0f / sum;
            for (int s = lane; s < sPad; s += 32) {
                float p = (s <= t) ? Sw[r * Tt + s] * inv : 0.0f;
                Pw[r * LVP + s] = f2bf(p);
            }
        }

        // ---- O[16x64] = P[16 x sPad] @ V[sPad x 64] ----
        v8f o[4] = {};
        const int kSteps = sPad >> 5;
        for (int ks = 0; ks < kSteps; ++ks) {
            int sBase = ks * 32;
            v16bf aP = ldfrag(&Pw[l16 * LVP + sBase + klo]);
            #pragma unroll
            for (int nt = 0; nt < 4; ++nt) {
                v16bf bV = ldfrag(&Vt[(nt * 16 + l16) * LVP + sBase + klo]);
                o[nt] = wmma_bf16(aP, bV, o[nt]);
            }
        }

        // ---- store O (bf16, concat-head layout [B,T,H*D]) ----
        #pragma unroll
        for (int nt = 0; nt < 4; ++nt) {
            int d = nt * 16 + l16;
            #pragma unroll
            for (int r = 0; r < 8; ++r) {
                int t = t0 + r + klo;
                attnb[base + (size_t)t * Cc + d] = f2bf(o[nt][r]);
            }
        }
    }
}

// ---------------------------------------------------------------------------
// out = x + LayerNorm(branch) * g + b  (norm on the branch, per reference).
// One wave per row of 384; 8 rows per 256-thread block.
// ---------------------------------------------------------------------------
__global__ __launch_bounds__(256) void ln_residual_kernel(
    const float* __restrict__ x, const float* __restrict__ br,
    const float* __restrict__ g, const float* __restrict__ bb,
    float* __restrict__ outf, bf16* __restrict__ outb)
{
    const int w = threadIdx.x >> 5;
    const int lane = threadIdx.x & 31;
    const size_t row = (size_t)blockIdx.x * 8 + w;
    const float* pr = br + row * Cc;

    float s = 0.0f;
    for (int c = lane; c < Cc; c += 32) s += pr[c];
    #pragma unroll
    for (int off = 16; off; off >>= 1) s += __shfl_xor(s, off, 32);
    float mean = s * (1.0f / Cc);

    float v = 0.0f;
    for (int c = lane; c < Cc; c += 32) { float d = pr[c] - mean; v += d * d; }
    #pragma unroll
    for (int off = 16; off; off >>= 1) v += __shfl_xor(v, off, 32);
    float rstd = rsqrtf(v * (1.0f / Cc) + 1e-5f);

    for (int c = lane; c < Cc; c += 32) {
        float val = x[row * Cc + c] + (pr[c] - mean) * rstd * g[c] + bb[c];
        outf[row * Cc + c] = val;
        if (outb) outb[row * Cc + c] = f2bf(val);
    }
}

// ---------------------------------------------------------------------------
// Host-side orchestration
// ---------------------------------------------------------------------------
extern "C" void kernel_launch(void* const* d_in, const int* in_sizes, int n_in,
                              void* d_out, int out_size, void* d_ws, size_t ws_size,
                              hipStream_t stream) {
    (void)in_sizes; (void)n_in; (void)out_size; (void)ws_size;

    const float* x      = (const float*)d_in[0];
    const float* q_w    = (const float*)d_in[1];
    const float* q_b    = (const float*)d_in[2];
    const float* k_w    = (const float*)d_in[3];
    const float* k_b    = (const float*)d_in[4];
    const float* v_w    = (const float*)d_in[5];
    const float* v_b    = (const float*)d_in[6];
    const float* proj_w = (const float*)d_in[7];
    const float* proj_b = (const float*)d_in[8];
    const float* ln1_g  = (const float*)d_in[9];
    const float* ln1_b  = (const float*)d_in[10];
    const float* w1     = (const float*)d_in[11];
    const float* b1     = (const float*)d_in[12];
    const float* w2     = (const float*)d_in[13];
    const float* b2     = (const float*)d_in[14];
    const float* ln2_g  = (const float*)d_in[15];
    const float* ln2_b  = (const float*)d_in[16];
    float* out = (float*)d_out;

    // --- workspace carve-up ---
    char* ws = (char*)d_ws;
    size_t off = 0;
    auto alloc = [&](size_t bytes) {
        char* p = ws + off;
        off += (bytes + 255) & ~(size_t)255;
        return p;
    };
    bf16* xb     = (bf16*)alloc((size_t)Mm * Cc * 2);
    bf16* qwb    = (bf16*)alloc((size_t)Cc * Cc * 2);
    bf16* kwb    = (bf16*)alloc((size_t)Cc * Cc * 2);
    bf16* vwb    = (bf16*)alloc((size_t)Cc * Cc * 2);
    bf16* pwb    = (bf16*)alloc((size_t)Cc * Cc * 2);
    bf16* w1b    = (bf16*)alloc((size_t)Cc * FFf * 2);
    bf16* w2b    = (bf16*)alloc((size_t)FFf * Cc * 2);
    bf16* qact   = (bf16*)alloc((size_t)Mm * Cc * 2);
    bf16* kact   = (bf16*)alloc((size_t)Mm * Cc * 2);
    bf16* vact   = (bf16*)alloc((size_t)Mm * Cc * 2);
    bf16* attnb  = (bf16*)alloc((size_t)Mm * Cc * 2);
    float* projf = (float*)alloc((size_t)Mm * Cc * 4);
    float* x1f   = (float*)alloc((size_t)Mm * Cc * 4);
    bf16* x1b    = (bf16*)alloc((size_t)Mm * Cc * 2);
    bf16* hb     = (bf16*)alloc((size_t)Mm * FFf * 2);
    float* fff   = (float*)alloc((size_t)Mm * Cc * 4);

    // --- 1. precision conversion + weight packing ---
    cvt_f32_bf16_kernel<<<4096, 256, 0, stream>>>(x, xb, Mm * Cc);
    pack_hcd_kernel<<<576, 256, 0, stream>>>(q_w, qwb);
    pack_hcd_kernel<<<576, 256, 0, stream>>>(k_w, kwb);
    pack_hcd_kernel<<<576, 256, 0, stream>>>(v_w, vwb);
    cvt_f32_bf16_kernel<<<576, 256, 0, stream>>>(proj_w, pwb, Cc * Cc);
    cvt_f32_bf16_kernel<<<2304, 256, 0, stream>>>(w1, w1b, Cc * FFf);
    cvt_f32_bf16_kernel<<<2304, 256, 0, stream>>>(w2, w2b, FFf * Cc);

    // --- 2. QKV projections (output [B,T,H*D] bf16, n = h*64+d) ---
    dim3 gQKV(Cc / GBN, Mm / GBM);
    gemm_bf16_kernel<0, 0, 1><<<gQKV, 256, 0, stream>>>(xb, qwb, q_b,
                                                        nullptr, qact, Mm, Cc, Cc);
    gemm_bf16_kernel<0, 0, 1><<<gQKV, 256, 0, stream>>>(xb, kwb, k_b,
                                                        nullptr, kact, Mm, Cc, Cc);
    gemm_bf16_kernel<0, 0, 1><<<gQKV, 256, 0, stream>>>(xb, vwb, v_b,
                                                        nullptr, vact, Mm, Cc, Cc);

    // --- 3. fused attention ---
    size_t attnLds = (size_t)(Tt * LKQ) * 2 * 2    // Kl + Ql
                   + (size_t)(Dd * LVP) * 2        // Vt
                   + (size_t)(8 * 16 * Tt) * 4     // S (f32, per wave)
                   + (size_t)(8 * 16 * LVP) * 2;   // P (bf16, per wave)
    attn_kernel<<<Bb * Hh, 256, attnLds, stream>>>(qact, kact, vact, attnb);

    // --- 4. output projection + residual LN1 ---
    gemm_bf16_kernel<0, 1, 0><<<gQKV, 256, 0, stream>>>(attnb, pwb, proj_b,
                                                        projf, nullptr, Mm, Cc, Cc);
    ln_residual_kernel<<<Mm / 8, 256, 0, stream>>>(x, projf, ln1_g, ln1_b,
                                                   x1f, x1b);

    // --- 5. FFN + residual LN2 ---
    dim3 gF1(FFf / GBN, Mm / GBM);
    gemm_bf16_kernel<1, 0, 1><<<gF1, 256, 0, stream>>>(x1b, w1b, b1,
                                                       nullptr, hb, Mm, FFf, Cc);
    dim3 gF2(Cc / GBN, Mm / GBM);
    gemm_bf16_kernel<0, 1, 0><<<gF2, 256, 0, stream>>>(hb, w2b, b2,
                                                       fff, nullptr, Mm, Cc, FFf);
    ln_residual_kernel<<<Mm / 8, 256, 0, stream>>>(x1f, fff, ln2_g, ln2_b,
                                                   out, nullptr);
}
